// sLSTMBlock_6030134084214
// MI455X (gfx1250) — matrix operand order repaired and checked by reading, not compiled, not run on hardware
//
#include <hip/hip_runtime.h>
#include <hip/hip_bf16.h>

// ---------------------------------------------------------------------------
// Types for CDNA5 WMMA (wave32): v16bf A/B fragments, v8f accumulator
// ---------------------------------------------------------------------------
typedef __attribute__((ext_vector_type(16))) __bf16 v16bf;
typedef __attribute__((ext_vector_type(8)))  float  v8f;
typedef __attribute__((ext_vector_type(8)))  unsigned short u16x8;

union Frag {
    v16bf v;
    u16x8 h[2];
};

__device__ __forceinline__ unsigned short f2bf(float f) {
    unsigned int u = __float_as_uint(f);
    u += 0x7FFFu + ((u >> 16) & 1u);   // round-to-nearest-even
    return (unsigned short)(u >> 16);
}
__device__ __forceinline__ float bf2f(unsigned short h) {
    return __uint_as_float(((unsigned int)h) << 16);
}

// ---------------------------------------------------------------------------
// CDNA5 async global->LDS copy (tracked by ASYNCcnt), 16 bytes per lane.
// ldsOff = raw LDS byte offset (addrspace(3)); g = per-lane global address.
// ---------------------------------------------------------------------------
__device__ __forceinline__ void async_copy_b128(unsigned ldsOff, const unsigned short* g) {
    asm volatile("global_load_async_to_lds_b128 %0, %1, off"
                 :: "v"(ldsOff), "v"((unsigned long long)(uintptr_t)g)
                 : "memory");
}

template <int N>
__device__ __forceinline__ void wait_asynccnt() {
#if __has_builtin(__builtin_amdgcn_s_wait_asynccnt)
    __builtin_amdgcn_s_wait_asynccnt(N);
#else
    asm volatile("s_wait_asynccnt %0" :: "i"(N) : "memory");
#endif
}

__device__ __forceinline__ unsigned lds_offset(const void* p) {
    // Shared-aperture flat addresses carry the LDS byte offset in the low 32 bits.
    return (unsigned)(uintptr_t)p;
}

// ---------------------------------------------------------------------------
// fp32 -> bf16 conversion
// ---------------------------------------------------------------------------
__global__ void cvt_bf16_kernel(const float* __restrict__ in,
                                unsigned short* __restrict__ out, int n) {
    int i = blockIdx.x * blockDim.x + threadIdx.x;
    if (i < n) out[i] = f2bf(in[i]);
}

// ---------------------------------------------------------------------------
// WMMA GEMM: C[M,N] = A[M,K] @ W[N,K]^T (+ bias[N]), A/W bf16, acc fp32.
// Workgroup tile 128x128, 256 threads = 8 waves, wave tile 32x64 (2x4 WMMA).
// K stepped in chunks of 32, double-buffered LDS filled by async copies:
// fill(k+1) overlaps the 8 WMMAs of chunk k. Requires M,N %128==0, K%32==0.
// ---------------------------------------------------------------------------
template <bool OUT_BF16, bool HAS_BIAS>
__global__ __launch_bounds__(256)
void wmma_gemm_kernel(const unsigned short* __restrict__ A,
                      const unsigned short* __restrict__ W,
                      const float* __restrict__ bias,
                      void* __restrict__ C,
                      int M, int N, int K) {
    __shared__ __attribute__((aligned(16))) unsigned short lA[2][128 * 32];
    __shared__ __attribute__((aligned(16))) unsigned short lB[2][128 * 32];

    const int tid  = threadIdx.x;
    const int lane = tid & 31;
    const int wv   = tid >> 5;          // 0..7
    const int wm   = (wv >> 1) * 32;    // wave M offset within 128 tile
    const int wn   = (wv & 1) * 64;     // wave N offset within 128 tile
    const int lm   = lane & 15;         // row/col within 16x16 tile
    const int lk8  = (lane >> 4) * 8;   // A-frag K sub-offset
    const int lk16 = (lane >> 4) * 16;  // unused (B frags use lk8, like A)

    const int mBase = blockIdx.y * 128;
    const int nBase = blockIdx.x * 128;

    // Per-thread staging slice: 16 bf16 (two b128 async copies) per tile fill
    const int ldRow  = tid >> 1;        // 0..127
    const int ldColH = (tid & 1) * 16;  // 0 or 16

    const unsigned short* gAbase = A + (size_t)(mBase + ldRow) * (size_t)K + ldColH;
    const unsigned short* gBbase = W + (size_t)(nBase + ldRow) * (size_t)K + ldColH;
    const unsigned sA0 = lds_offset(&lA[0][ldRow * 32 + ldColH]);
    const unsigned sB0 = lds_offset(&lB[0][ldRow * 32 + ldColH]);
    const unsigned bufBytes = 128 * 32 * 2;

    auto issue_fill = [&](int buf, int k0) {
        const unsigned short* ga = gAbase + k0;
        const unsigned short* gb = gBbase + k0;
        const unsigned da = sA0 + (unsigned)buf * bufBytes;
        const unsigned db = sB0 + (unsigned)buf * bufBytes;
        async_copy_b128(da,      ga);       // A: 4 async instrs per wave/fill
        async_copy_b128(da + 16, ga + 8);
        async_copy_b128(db,      gb);
        async_copy_b128(db + 16, gb + 8);
    };

    const v8f zero = {0.f, 0.f, 0.f, 0.f, 0.f, 0.f, 0.f, 0.f};
    v8f acc[2][4];
#pragma unroll
    for (int mi = 0; mi < 2; ++mi)
#pragma unroll
        for (int ni = 0; ni < 4; ++ni) acc[mi][ni] = zero;

    const int nIter = K / 32;
    issue_fill(0, 0);

    for (int i = 0; i < nIter; ++i) {
        const int cur = i & 1;
        if (i + 1 < nIter) {
            issue_fill(cur ^ 1, (i + 1) * 32);  // overlaps this chunk's WMMAs
            wait_asynccnt<4>();   // own 4 current-buffer copies done (in-order)
        } else {
            wait_asynccnt<0>();
        }
        __syncthreads();          // all waves' current-buffer copies visible

        // A fragments: lane holds row (wm+mi*16+lm), K = lk8..+7 and 16+lk8..+7
        Frag af[2];
#pragma unroll
        for (int mi = 0; mi < 2; ++mi) {
            const int row = wm + mi * 16 + lm;
            af[mi].h[0] = *(const u16x8*)&lA[cur][row * 32 + lk8];
            af[mi].h[1] = *(const u16x8*)&lA[cur][row * 32 + 16 + lk8];
        }
        // B fragments: lane holds col (wn+ni*16+lm), K = lk8..+7 and 16+lk8..+7
        Frag bfr[4];
#pragma unroll
        for (int ni = 0; ni < 4; ++ni) {
            const int col = wn + ni * 16 + lm;
            bfr[ni].h[0] = *(const u16x8*)&lB[cur][col * 32 + lk8];
            bfr[ni].h[1] = *(const u16x8*)&lB[cur][col * 32 + 16 + lk8];
        }
#pragma unroll
        for (int mi = 0; mi < 2; ++mi)
#pragma unroll
            for (int ni = 0; ni < 4; ++ni)
                acc[mi][ni] = __builtin_amdgcn_wmma_f32_16x16x32_bf16(
                    false, af[mi].v, false, bfr[ni].v,
                    (short)0, acc[mi][ni], false, false);

        __syncthreads();  // all reads of buf[cur] done before iter i+1 refills it
    }

    // Epilogue: C/D layout -> lane (n = lm), VGPR r -> M = (lane>>4)*8 + r
    const int mOff = (lane >> 4) * 8;
#pragma unroll
    for (int mi = 0; mi < 2; ++mi) {
#pragma unroll
        for (int ni = 0; ni < 4; ++ni) {
            const int col = nBase + wn + ni * 16 + lm;
            float bv = 0.f;
            if (HAS_BIAS) bv = bias[col];
#pragma unroll
            for (int r = 0; r < 8; ++r) {
                const int row = mBase + wm + mi * 16 + mOff + r;
                const float v = acc[mi][ni][r] + bv;
                if (OUT_BF16)
                    ((unsigned short*)C)[(size_t)row * (size_t)N + col] = f2bf(v);
                else
                    ((float*)C)[(size_t)row * (size_t)N + col] = v;
            }
        }
    }
}

// ---------------------------------------------------------------------------
// Sequential sLSTM scan: 16 blocks (B*HEADS), 512 threads (one per HD elem).
// gates: (B*L, 8192) bf16, laid out n = h*2048 + {i:0,f:512,o:1024,c:1536}+d
// hout : (B*L, 2048) bf16, hidden idx = h*512 + d
// ---------------------------------------------------------------------------
__global__ __launch_bounds__(512)
void slstm_scan_kernel(const unsigned short* __restrict__ gates,
                       const float* __restrict__ rms_w,
                       unsigned short* __restrict__ hout) {
    const int b = blockIdx.x >> 2;   // 0..3
    const int h = blockIdx.x & 3;    // 0..3
    const int d = threadIdx.x;       // 0..511
    const int lane = d & 31;
    const int wv = d >> 5;           // 0..15

    __shared__ float partial[16];

    float c = 0.f;
    const float w = rms_w[d];

    for (int l = 0; l < 2048; ++l) {
        const size_t base = ((size_t)(b * 2048 + l)) * 8192 + (size_t)h * 2048 + d;
        const float gi = bf2f(gates[base]);
        const float gf = bf2f(gates[base + 512]);
        const float go = bf2f(gates[base + 1024]);
        const float gc = bf2f(gates[base + 1536]);

        const float ig = __expf(gi);
        const float fg = 1.f / (1.f + __expf(-gf));
        const float og = 1.f / (1.f + __expf(-go));
        const float cd = tanhf(gc);

        c = fg * c + ig * cd;

        // RMS over HD=512: wave32 reduce then 16-partial combine
        float ss = c * c;
#pragma unroll
        for (int off = 16; off > 0; off >>= 1) ss += __shfl_xor(ss, off, 32);
        if (lane == 0) partial[wv] = ss;
        __syncthreads();
        float tot = 0.f;
#pragma unroll
        for (int i = 0; i < 16; ++i) tot += partial[i];
        const float cn = c * rsqrtf(tot * (1.f / 512.f) + 1e-6f) * w;
        const float hv = og * tanhf(cn);

        hout[((size_t)(b * 2048 + l)) * 2048 + (size_t)h * 512 + d] = f2bf(hv);
        __syncthreads();  // protect partial[] for next step
    }
}

// ---------------------------------------------------------------------------
// Launch
// ---------------------------------------------------------------------------
extern "C" void kernel_launch(void* const* d_in, const int* in_sizes, int n_in,
                              void* d_out, int out_size, void* d_ws, size_t ws_size,
                              hipStream_t stream) {
    (void)in_sizes; (void)n_in; (void)out_size; (void)ws_size;

    const int B = 4, L = 2048, DIM = 2048, HIDDEN = 2048;
    const int M = B * L;               // 8192
    const int NG = 4 * HIDDEN;         // 8192

    const float* x      = (const float*)d_in[0];  // (B,L,DIM)
    const float* W_in   = (const float*)d_in[1];  // (HIDDEN,DIM)
    const float* W_gate = (const float*)d_in[2];  // (4H,HIDDEN)
    const float* b_gate = (const float*)d_in[3];  // (4H)
    const float* rms_w  = (const float*)d_in[4];  // (512)
    const float* W_out  = (const float*)d_in[5];  // (DIM,HIDDEN)
    float* out = (float*)d_out;

    char* ws = (char*)d_ws;
    unsigned short* xb   = (unsigned short*)ws; ws += (size_t)M * DIM * 2;       // x bf16
    unsigned short* wib  = (unsigned short*)ws; ws += (size_t)HIDDEN * DIM * 2;  // W_in bf16
    unsigned short* xpb  = (unsigned short*)ws; ws += (size_t)M * HIDDEN * 2;    // xp bf16
    unsigned short* wgb  = (unsigned short*)ws; ws += (size_t)NG * HIDDEN * 2;   // W_gate bf16
    unsigned short* gtb  = (unsigned short*)ws; ws += (size_t)M * NG * 2;        // gates bf16
    unsigned short* hb   = (unsigned short*)ws; ws += (size_t)M * HIDDEN * 2;    // h bf16
    unsigned short* wob  = (unsigned short*)ws;                                  // W_out bf16

    const int nx = M * DIM, nwi = HIDDEN * DIM, nwg = NG * HIDDEN, nwo = DIM * HIDDEN;
    cvt_bf16_kernel<<<(nx  + 255) / 256, 256, 0, stream>>>(x,      xb,  nx);
    cvt_bf16_kernel<<<(nwi + 255) / 256, 256, 0, stream>>>(W_in,   wib, nwi);
    cvt_bf16_kernel<<<(nwg + 255) / 256, 256, 0, stream>>>(W_gate, wgb, nwg);
    cvt_bf16_kernel<<<(nwo + 255) / 256, 256, 0, stream>>>(W_out,  wob, nwo);

    // GEMM1: xp = x @ W_in^T  -> bf16
    wmma_gemm_kernel<true, false><<<dim3(HIDDEN / 128, M / 128), 256, 0, stream>>>(
        xb, wib, nullptr, xpb, M, HIDDEN, DIM);
    // GEMM2: gates = xp @ W_gate^T + b  -> bf16
    wmma_gemm_kernel<true, true><<<dim3(NG / 128, M / 128), 256, 0, stream>>>(
        xpb, wgb, b_gate, gtb, M, NG, HIDDEN);
    // Sequential recurrence
    slstm_scan_kernel<<<16, 512, 0, stream>>>(gtb, rms_w, hb);
    // GEMM3: out = h @ W_out^T  -> fp32
    wmma_gemm_kernel<false, false><<<dim3(DIM / 128, M / 128), 256, 0, stream>>>(
        hb, wob, nullptr, out, M, DIM, HIDDEN);
}
